// Box3dAttention_33457795236458
// MI455X (gfx1250) — matrix operand order repaired and checked by reading, not compiled, not run on hardware
//
#include <hip/hip_runtime.h>
#include <hip/hip_bf16.h>
#include <math.h>

// ---------------------------------------------------------------------------
// Box3dAttention forward for gfx1250 (MI455X).
// GEMMs: v_wmma_f32_16x16x32_bf16; one wave computes a 16x64 C strip with a
// branch-free k-loop (tails handled by index clamping, not control flow).
// ---------------------------------------------------------------------------

#define D_MODEL   256
#define NUM_HEAD  8
#define NUM_LEVEL 4
#define KS        5
#define NUM_POINT 25
#define HEAD_DIM  32
#define LP        (NUM_LEVEL * NUM_POINT)   // 100

typedef __bf16 bf16x16 __attribute__((ext_vector_type(16)));
typedef float  f32x8   __attribute__((ext_vector_type(8)));

__device__ __forceinline__ __bf16 f2bf(float f) {
    return (__bf16)f;   // native v_cvt_pk_bf16_f32 on gfx1250
}

__device__ __forceinline__ bf16x16 cvt16(float4 a0, float4 a1,
                                         float4 a2, float4 a3) {
    bf16x16 r;
    r[0]  = f2bf(a0.x); r[1]  = f2bf(a0.y);
    r[2]  = f2bf(a0.z); r[3]  = f2bf(a0.w);
    r[4]  = f2bf(a1.x); r[5]  = f2bf(a1.y);
    r[6]  = f2bf(a1.z); r[7]  = f2bf(a1.w);
    r[8]  = f2bf(a2.x); r[9]  = f2bf(a2.y);
    r[10] = f2bf(a2.z); r[11] = f2bf(a2.w);
    r[12] = f2bf(a3.x); r[13] = f2bf(a3.y);
    r[14] = f2bf(a3.z); r[15] = f2bf(a3.w);
    return r;
}

// ---------------------------------------------------------------------------
// GEMM:  C[M,N] = A[M,K] * W[N,K]^T + bias[N]
// A row-major f32, W row-major (N,K) f32, C row-major f32.
// K multiple of 32, N multiple of 16, M guarded (stores only).
// One wave per 16(M) x 64(N) strip: 4 WMMAs per k-step share one A fragment.
// Tail handling by clamping (redundant compute, guarded stores) keeps the
// k-loop branch-free so accumulators stay pinned in even-aligned VGPR blocks.
// ---------------------------------------------------------------------------
__global__ __launch_bounds__(32)
void box3d_wmma_gemm(const float* __restrict__ A,
                     const float* __restrict__ W,
                     const float* __restrict__ bias,
                     float* __restrict__ C,
                     int M, int N, int K)
{
    const int tileM  = blockIdx.x;
    const int tileN0 = blockIdx.y * 4;     // first of up to 4 N-tiles
    const int lane   = threadIdx.x & 31;
    const int half   = lane >> 4;          // 0: lanes 0-15, 1: lanes 16-31
    const int lm     = lane & 15;

    const int nTilesTotal = N / 16;
    // Clamp A row: rows >= M redundantly recompute row M-1; their outputs
    // are never stored (each C row is an independent dot product).
    const int row  = tileM * 16 + lm;
    const int rowc = row < M ? row : (M - 1);

    // Clamp N tiles: tail blocks duplicate the last tile; never stored.
    int tn[4];
#pragma unroll
    for (int t = 0; t < 4; ++t) {
        int x = tileN0 + t;
        tn[t] = x < nTilesTotal ? x : (nTilesTotal - 1);
    }

    f32x8 acc[4];
#pragma unroll
    for (int t = 0; t < 4; ++t)
#pragma unroll
        for (int r = 0; r < 8; ++r) acc[t][r] = 0.0f;

    const float* aRow = A + (size_t)rowc * K;

    for (int k0 = 0; k0 < K; k0 += 32) {
        // ----- A fragment: 16-bit A 16x32 layout (ISA 7.12.2) -----
        // lane half 0: K = {k0+0..7,  k0+16..23}
        // lane half 1: K = {k0+8..15, k0+24..31}
        const float* ap = aRow + k0 + half * 8;
        const bf16x16 afrag = cvt16(*(const float4*)(ap + 0),
                                    *(const float4*)(ap + 4),
                                    *(const float4*)(ap + 16),
                                    *(const float4*)(ap + 20));

        // ----- 4 B fragments / 4 independent WMMAs share the A fragment ----
        // B is 32x16 = W[col][k]^T; lanes 0-15 hold K = {k0+0..7, k0+16..23},
        // lanes 16-31 hold K = {k0+8..15, k0+24..31}.
#pragma unroll
        for (int t = 0; t < 4; ++t) {
            const int col = tn[t] * 16 + lm;
            const float* wp = W + (size_t)col * K + k0 + half * 8;
            const bf16x16 bfrag = cvt16(*(const float4*)(wp + 0),
                                        *(const float4*)(wp + 4),
                                        *(const float4*)(wp + 16),
                                        *(const float4*)(wp + 20));
            // (neg_a, A, neg_b, B, c_mod, C, reuse_a, reuse_b)
            acc[t] = __builtin_amdgcn_wmma_f32_16x16x32_bf16(
                false, afrag, false, bfrag, (short)0, acc[t], false, false);
        }
    }

    // ----- store: C/D layout: lane => N=lm, vgpr r => M = half*8 + r -----
#pragma unroll
    for (int t = 0; t < 4; ++t) {
        if (tileN0 + t < nTilesTotal) {
            const int col = tn[t] * 16 + lm;
            const float bv = bias ? bias[col] : 0.0f;
#pragma unroll
            for (int r = 0; r < 8; ++r) {
                int mrow = tileM * 16 + half * 8 + r;
                if (mrow < M)
                    C[(size_t)mrow * N + col] = acc[t][r] + bv;
            }
        }
    }
}

// ---------------------------------------------------------------------------
// Softmax over 100 contiguous elements per row; one wave32 per row.
// ---------------------------------------------------------------------------
__global__ __launch_bounds__(128)
void box3d_softmax100(const float* __restrict__ in, float* __restrict__ out,
                      int rows)
{
    const int wave = blockIdx.x * 4 + (threadIdx.x >> 5);
    const int lane = threadIdx.x & 31;
    if (wave >= rows) return;

    const float* rp = in + (size_t)wave * LP;
    float*       wp = out + (size_t)wave * LP;

    float v[4];
#pragma unroll
    for (int t = 0; t < 4; ++t) {
        int j = lane + 32 * t;
        v[t] = (j < LP) ? rp[j] : -INFINITY;
    }
    float m = fmaxf(fmaxf(v[0], v[1]), fmaxf(v[2], v[3]));
#pragma unroll
    for (int off = 16; off > 0; off >>= 1)
        m = fmaxf(m, __shfl_xor(m, off, 32));

    float e[4], s = 0.0f;
#pragma unroll
    for (int t = 0; t < 4; ++t) {
        int j = lane + 32 * t;
        e[t] = (j < LP) ? __expf(v[t] - m) : 0.0f;
        s += e[t];
    }
#pragma unroll
    for (int off = 16; off > 0; off >>= 1)
        s += __shfl_xor(s, off, 32);
    const float inv = 1.0f / s;

#pragma unroll
    for (int t = 0; t < 4; ++t) {
        int j = lane + 32 * t;
        if (j < LP) wp[j] = e[t] * inv;
    }
}

// ---------------------------------------------------------------------------
// Sampling: one wave per (b, q, h); lane = head-dim channel (each corner
// gather is one contiguous 128B line; v_proj is L2-resident: 92MB < 192MB).
// ---------------------------------------------------------------------------
__global__ __launch_bounds__(128)
void box3d_sample(const float* __restrict__ v,        // (B, LV, 256) projected
                  const float* __restrict__ aw,       // (B*LQ*H, 100) softmaxed
                  const float* __restrict__ ob,       // (B*LQ, 160) box head
                  const float* __restrict__ ref_w,    // (B*LQ, 7)
                  const float* __restrict__ ratios,   // (B, 4, 2)
                  const int*   __restrict__ v_shape,  // (4, 2)
                  const int*   __restrict__ v_start,  // (4,)
                  const unsigned char* __restrict__ v_mask, // (B, LV)
                  float* __restrict__ acc_out,        // (B*LQ, 256)
                  int B, int LQ, int LV, int nWaves)
{
    const int wave = blockIdx.x * 4 + (threadIdx.x >> 5);
    const int lane = threadIdx.x & 31;
    if (wave >= nWaves) return;

    const int h  = wave % NUM_HEAD;
    const int bq = wave / NUM_HEAD;
    const int b  = bq / LQ;

    // Pull the per-query scalar tables toward the WGP early
    __builtin_prefetch(ob + (size_t)bq * 160 + h * 20, 0, 3);
    __builtin_prefetch(aw + (size_t)wave * LP, 0, 3);

    const float rw0 = ref_w[(size_t)bq * 7 + 0];
    const float rw1 = ref_w[(size_t)bq * 7 + 1];
    const float rw3 = ref_w[(size_t)bq * 7 + 3];
    const float rw4 = ref_w[(size_t)bq * 7 + 4];
    const float rw6 = ref_w[(size_t)bq * 7 + 6];

    float acc = 0.0f;

    for (int l = 0; l < NUM_LEVEL; ++l) {
        const int Hl = v_shape[2 * l + 0];
        const int Wl = v_shape[2 * l + 1];
        const int s  = v_start[l];

        const float* obp = ob + (size_t)bq * 160 + h * 20 + l * 5;
        const float o0 = obp[0], o1 = obp[1], o2 = obp[2], o3 = obp[3],
                    o4 = obp[4];

        const float rx = ratios[(b * NUM_LEVEL + l) * 2 + 0];
        const float ry = ratios[(b * NUM_LEVEL + l) * 2 + 1];

        const float ang = (rw6 + o4 * 0.0625f) * 6.283185307179586f;
        const float ca = __cosf(ang), sa = __sinf(ang);

        const float cx = rw0 + o0 * 0.125f * rw3;
        const float cy = rw1 + o1 * 0.125f * rw4;
        const float sw = fmaxf(rw3 + o2 * 0.125f * rw3, 0.0f);
        const float sh = fmaxf(rw4 + o3 * 0.125f * rw4, 0.0f);

        const float* awp = aw + (size_t)wave * LP + l * NUM_POINT;

        for (int p = 0; p < NUM_POINT; ++p) {
            const float kx = (float)((p % KS) - 2) * 0.2f;
            const float ky = (float)((p / KS) - 2) * 0.2f;
            const float gx = kx * sw;
            const float gy = ky * sh;
            const float lx_ = (cx + gx * ca - gy * sa) * rx;
            const float ly_ = (cy + gx * sa + gy * ca) * ry;

            const float x = lx_ * (float)Wl - 0.5f;
            const float y = ly_ * (float)Hl - 0.5f;
            const float x0f = floorf(x), y0f = floorf(y);
            const float fx = x - x0f, fy = y - y0f;
            const int x0 = (int)x0f, y0 = (int)y0f;

            const float wgt = awp[p];

            float samp = 0.0f;
#pragma unroll
            for (int c = 0; c < 4; ++c) {
                const int yy = y0 + (c >> 1);
                const int xx = x0 + (c & 1);
                const float wx = (c & 1) ? fx : (1.0f - fx);
                const float wy = (c >> 1) ? fy : (1.0f - fy);
                bool valid = (yy >= 0) && (yy < Hl) && (xx >= 0) && (xx < Wl);
                int cyy = yy < 0 ? 0 : (yy > Hl - 1 ? Hl - 1 : yy);
                int cxx = xx < 0 ? 0 : (xx > Wl - 1 ? Wl - 1 : xx);
                const size_t lin = (size_t)(s + cyy * Wl + cxx);
                float val =
                    v[((size_t)b * LV + lin) * D_MODEL + h * HEAD_DIM + lane];
                if (v_mask[(size_t)b * LV + lin]) valid = false;
                samp += valid ? (val * wx * wy) : 0.0f;
            }
            acc += wgt * samp;
        }
    }

    acc_out[(size_t)bq * D_MODEL + h * HEAD_DIM + lane] = acc;
}

// ---------------------------------------------------------------------------

static inline void launch_gemm(const float* A, const float* W, const float* b,
                               float* C, int M, int N, int K,
                               hipStream_t stream)
{
    dim3 grid((M + 15) / 16, (N / 16 + 3) / 4);
    box3d_wmma_gemm<<<grid, 32, 0, stream>>>(A, W, b, C, M, N, K);
}

extern "C" void kernel_launch(void* const* d_in, const int* in_sizes, int n_in,
                              void* d_out, int out_size, void* d_ws,
                              size_t ws_size, hipStream_t stream)
{
    (void)n_in; (void)out_size; (void)ws_size;

    const float*         query   = (const float*)d_in[0];
    const float*         value   = (const float*)d_in[1];
    const int*           v_shape = (const int*)d_in[2];
    const unsigned char* v_mask  = (const unsigned char*)d_in[3];
    const int*           v_start = (const int*)d_in[4];
    const float*         ratios  = (const float*)d_in[5];
    const float*         ref_w   = (const float*)d_in[6];
    const float*         vp_w    = (const float*)d_in[7];
    const float*         vp_b    = (const float*)d_in[8];
    const float*         op_w    = (const float*)d_in[9];
    const float*         op_b    = (const float*)d_in[10];
    const float*         box_w   = (const float*)d_in[11];
    const float*         box_b   = (const float*)d_in[12];
    const float*         attn_w  = (const float*)d_in[13];
    const float*         attn_b  = (const float*)d_in[14];

    // Derive sizes: v_valid_ratios = (B,1,1,4,1,2); value = (B,LV,256);
    // query = (B,LQ,256).
    const int B   = in_sizes[5] / (NUM_LEVEL * 2);
    const int BLV = in_sizes[1] / D_MODEL;
    const int LV  = BLV / B;
    const int BLQ = in_sizes[0] / D_MODEL;
    const int LQ  = BLQ / B;

    float* ws     = (float*)d_ws;
    float* v_proj = ws;                               // B*LV*256
    float* ob     = v_proj + (size_t)BLV * D_MODEL;   // BLQ*160
    float* acc    = ob + (size_t)BLQ * 160;           // BLQ*256
    float* aw_raw = acc + (size_t)BLQ * D_MODEL;      // BLQ*800

    float* out0 = (float*)d_out;                      // (B,LQ,256)
    float* aw   = out0 + (size_t)BLQ * D_MODEL;       // (B,LQ,8,100) softmaxed

    // 1) value projection (dominant GEMM; ~11.8 GFLOP, bf16 WMMA)
    launch_gemm(value, vp_w, vp_b, v_proj, BLV, D_MODEL, D_MODEL, stream);

    // 2) attention logits (BLQ x 800)
    launch_gemm(query, attn_w, attn_b, aw_raw, BLQ,
                NUM_HEAD * LP, D_MODEL, stream);

    // 3) softmax over 100 -> second output region
    {
        const int rows = BLQ * NUM_HEAD;
        box3d_softmax100<<<(rows + 3) / 4, 128, 0, stream>>>(aw_raw, aw, rows);
    }

    // 4) box head (BLQ x 160)
    launch_gemm(query, box_w, box_b, ob, BLQ,
                NUM_LEVEL * NUM_HEAD * 5, D_MODEL, stream);

    // 5) deformable bilinear sampling -> acc (BLQ x 256)
    {
        const int nWaves = BLQ * NUM_HEAD;
        box3d_sample<<<(nWaves + 3) / 4, 128, 0, stream>>>(
            v_proj, aw, ob, ref_w, ratios, v_shape, v_start, v_mask, acc,
            B, LQ, LV, nWaves);
    }

    // 6) output projection -> first output region
    launch_gemm(acc, op_w, op_b, out0, BLQ, D_MODEL, D_MODEL, stream);
}
